// DynamicGCN_71382356459940
// MI455X (gfx1250) — compile-verified, matrix operand order checked by reading, not run on hardware
//
#include <hip/hip_runtime.h>

typedef __attribute__((ext_vector_type(16))) _Float16 v16h;
typedef __attribute__((ext_vector_type(8)))  _Float16 v8h;
typedef __attribute__((ext_vector_type(8)))  float    v8f;

#define D 128
#define EPS 1e-5f

// ---------------------------------------------------------------------------
// f32 -> f16 elementwise convert (x -> Xh)
// ---------------------------------------------------------------------------
__global__ __launch_bounds__(256) void cvt_f32_to_f16(
    const float* __restrict__ in, _Float16* __restrict__ out, int n)
{
    int i = blockIdx.x * 256 + threadIdx.x;
    if (i < n) out[i] = (_Float16)in[i];
}

// ---------------------------------------------------------------------------
// W (f32, row-major KxN) -> Wc (f16, column-major: Wc[n*128 + k])
// ---------------------------------------------------------------------------
__global__ __launch_bounds__(128) void prep_w(
    const float* __restrict__ W, _Float16* __restrict__ Wc)
{
    int k = threadIdx.x;   // 0..127
    int n = blockIdx.x;    // 0..127
    Wc[n * D + k] = (_Float16)W[k * D + n];
}

// ---------------------------------------------------------------------------
// H = relu(A @ B + bias)   A: [M,128] f16 row-major, B: col-major f16
// One wave per 16x128 row strip: A fragments loaded once (4 x v16h),
// reused across all 8 n-tiles -> 32 v_wmma_f32_16x16x32_f16 per wave.
// B (32 KB) is L2-resident and shared by all blocks.
//
// Fragment layout per CDNA5 ISA (16-bit 16x32): lane l holds row/col
// (l & 15); lanes 0-15 carry K = 0..7 and 16..23, lanes 16-31 carry
// K = 8..15 and 24..31  -> two aligned b128 loads per fragment.
// ---------------------------------------------------------------------------
__global__ __launch_bounds__(32) void gemm_relu_wmma(
    const _Float16* __restrict__ A, const _Float16* __restrict__ Bc,
    const float* __restrict__ bias, float* __restrict__ Out, int M)
{
    const int lane = threadIdx.x;
    const int r    = lane & 15;
    const int hi   = lane >> 4;
    const int row0 = blockIdx.x * 16;

    const _Float16* arow = A + (size_t)(row0 + r) * D + hi * 8;

    // Load the entire 16x128 A strip as 4 fragments, once.
    v16h a[4];
#pragma unroll
    for (int kk = 0; kk < 4; ++kk) {
        v8h al = *(const v8h*)(arow + kk * 32);
        v8h ah = *(const v8h*)(arow + kk * 32 + 16);
#pragma unroll
        for (int i = 0; i < 8; ++i) { a[kk][i] = al[i]; a[kk][8 + i] = ah[i]; }
    }

#pragma unroll
    for (int nt = 0; nt < 8; ++nt) {
        const int n0 = nt * 16;
        const _Float16* bcol = Bc + (size_t)(n0 + r) * D + hi * 8;

        v8f c = {};
#pragma unroll
        for (int kk = 0; kk < 4; ++kk) {
            v8h bl = *(const v8h*)(bcol + kk * 32);
            v8h bh = *(const v8h*)(bcol + kk * 32 + 16);
            v16h b;
#pragma unroll
            for (int i = 0; i < 8; ++i) { b[i] = bl[i]; b[8 + i] = bh[i]; }
            c = __builtin_amdgcn_wmma_f32_16x16x32_f16(
                    /*neg_a=*/false, a[kk], /*neg_b=*/false, b,
                    /*c_mod=*/(short)0, c, /*reuse_a=*/false, /*reuse_b=*/false);
        }

        const float bv = bias[n0 + r];
#pragma unroll
        for (int i = 0; i < 8; ++i) {
            float v = c[i] + bv;        // C/D layout: VGPR i -> M = i + 8*hi
            v = v > 0.f ? v : 0.f;
            Out[(size_t)(row0 + i + 8 * hi) * D + n0 + r] = v;
        }
    }
}

// ---------------------------------------------------------------------------
// Edge scatter: one wave per edge, float4 per lane (32*4 = 128 dims)
// ACC[dst] += ew * H[src]   (H, ACC are L2-resident: 25.6 MB each vs 192 MB L2)
// ---------------------------------------------------------------------------
__global__ __launch_bounds__(256) void scatter_add(
    const long long* __restrict__ src, const long long* __restrict__ dst,
    const float* __restrict__ ew, const float* __restrict__ H,
    float* __restrict__ ACC, int E)
{
    int e = blockIdx.x * 8 + (threadIdx.x >> 5);
    if (e >= E) return;
    int lane = threadIdx.x & 31;

    long long s = src[e];
    long long d = dst[e];
    float     w = ew[e];

    const float4 v = *(const float4*)(H + (size_t)s * D + lane * 4);
    float* a = ACC + (size_t)d * D + lane * 4;
    atomicAdd(a + 0, w * v.x);
    atomicAdd(a + 1, w * v.y);
    atomicAdd(a + 2, w * v.z);
    atomicAdd(a + 3, w * v.w);
}

// ---------------------------------------------------------------------------
// LayerNorm over (H + ACC), one wave per row, wave32 shuffle reduction.
// OutT = _Float16 (layer-1 intermediate, feeds next GEMM) or float (d_out).
// ---------------------------------------------------------------------------
template <typename OutT>
__global__ __launch_bounds__(256) void layernorm_k(
    const float* __restrict__ H, const float* __restrict__ ACC,
    const float* __restrict__ g, const float* __restrict__ be,
    OutT* __restrict__ out, int N)
{
    int row = blockIdx.x * 8 + (threadIdx.x >> 5);
    if (row >= N) return;
    int lane = threadIdx.x & 31;

    const size_t base = (size_t)row * D + lane * 4;
    float4 h = *(const float4*)(H + base);
    float4 a = *(const float4*)(ACC + base);
    float v0 = h.x + a.x, v1 = h.y + a.y, v2 = h.z + a.z, v3 = h.w + a.w;

    float s  = v0 + v1 + v2 + v3;
    float sq = v0 * v0 + v1 * v1 + v2 * v2 + v3 * v3;
#pragma unroll
    for (int o = 16; o > 0; o >>= 1) {
        s  += __shfl_xor(s,  o, 32);
        sq += __shfl_xor(sq, o, 32);
    }
    const float m    = s * (1.f / D);
    const float var  = sq * (1.f / D) - m * m;
    const float rstd = __frsqrt_rn(var + EPS);

    const int dd = lane * 4;
    out[(size_t)row * D + dd + 0] = (OutT)((v0 - m) * rstd * g[dd + 0] + be[dd + 0]);
    out[(size_t)row * D + dd + 1] = (OutT)((v1 - m) * rstd * g[dd + 1] + be[dd + 1]);
    out[(size_t)row * D + dd + 2] = (OutT)((v2 - m) * rstd * g[dd + 2] + be[dd + 2]);
    out[(size_t)row * D + dd + 3] = (OutT)((v3 - m) * rstd * g[dd + 3] + be[dd + 3]);
}

// ---------------------------------------------------------------------------
extern "C" void kernel_launch(void* const* d_in, const int* in_sizes, int n_in,
                              void* d_out, int out_size, void* d_ws, size_t ws_size,
                              hipStream_t stream)
{
    const float*     x   = (const float*)d_in[0];
    const long long* ei  = (const long long*)d_in[1];   // [2, E] int64
    const float*     ew  = (const float*)d_in[2];
    const float*     W1  = (const float*)d_in[3];
    const float*     b1  = (const float*)d_in[4];
    const float*     g1  = (const float*)d_in[5];
    const float*     be1 = (const float*)d_in[6];
    const float*     W2  = (const float*)d_in[7];
    const float*     b2  = (const float*)d_in[8];
    const float*     g2  = (const float*)d_in[9];
    const float*     be2 = (const float*)d_in[10];

    const int ND = in_sizes[0];     // N * 128
    const int N  = ND / D;          // 50000
    const int E  = in_sizes[2];     // 800000

    // Workspace layout (all 256B-aligned):
    //   [0)            W1h  f16 col-major  32 KB
    //   [32768)        W2h  f16 col-major  32 KB
    //   [65536)        Xh   f16 [N,128]    (reused as LN1 f16 output)
    //   [.. + 2*ND)    H    f32 [N,128]
    //   [.. + 4*ND)    ACC  f32 [N,128]
    char* ws = (char*)d_ws;
    _Float16* W1h = (_Float16*)ws;
    _Float16* W2h = (_Float16*)(ws + 32768);
    _Float16* Xh  = (_Float16*)(ws + 65536);
    float*    H   = (float*)(ws + 65536 + (size_t)ND * 2);
    float*    ACC = (float*)(ws + 65536 + (size_t)ND * 2 + (size_t)ND * 4);

    const long long* esrc = ei;
    const long long* edst = ei + E;

    const int gemm_blocks = N / 16;   // 3125 one-wave blocks, 16x128 strip each

    // --- prep ---
    cvt_f32_to_f16<<<(ND + 255) / 256, 256, 0, stream>>>(x, Xh, ND);
    prep_w<<<D, D, 0, stream>>>(W1, W1h);
    prep_w<<<D, D, 0, stream>>>(W2, W2h);

    // --- layer 1 ---
    gemm_relu_wmma<<<gemm_blocks, 32, 0, stream>>>(Xh, W1h, b1, H, N);
    hipMemsetAsync(ACC, 0, (size_t)ND * 4, stream);
    scatter_add<<<(E + 7) / 8, 256, 0, stream>>>(esrc, edst, ew, H, ACC, E);
    layernorm_k<_Float16><<<(N + 7) / 8, 256, 0, stream>>>(H, ACC, g1, be1, Xh, N);

    // --- layer 2 ---
    gemm_relu_wmma<<<gemm_blocks, 32, 0, stream>>>(Xh, W2h, b2, H, N);
    hipMemsetAsync(ACC, 0, (size_t)ND * 4, stream);
    scatter_add<<<(E + 7) / 8, 256, 0, stream>>>(esrc, edst, ew, H, ACC, E);
    layernorm_k<float><<<(N + 7) / 8, 256, 0, stream>>>(H, ACC, g2, be2, (float*)d_out, N);
}
